// Compositional_Attention_25451976196688
// MI455X (gfx1250) — matrix operand order repaired and checked by reading, not compiled, not run on hardware
//
#include <hip/hip_runtime.h>

typedef __bf16 bf16;
typedef __attribute__((ext_vector_type(16))) __bf16          v16bf;
typedef __attribute__((ext_vector_type(8)))  float           v8f;
typedef __attribute__((ext_vector_type(8)))  unsigned short  u16x8;
typedef __attribute__((ext_vector_type(16))) unsigned short  u16x16;
typedef __attribute__((ext_vector_type(4)))  unsigned int    u32x4;
typedef __attribute__((ext_vector_type(8)))  int             i32x8;
typedef __attribute__((ext_vector_type(4)))  int             i32x4;

// Assemble a 16-element bf16 WMMA operand from two 8-element (16B) chunks.
static __device__ __forceinline__ v16bf frag2(const bf16* p0, const bf16* p1) {
  u16x8 a = *(const u16x8*)p0;
  u16x8 b = *(const u16x8*)p1;
  u16x16 c = __builtin_shufflevector(a, b, 0,1,2,3,4,5,6,7,8,9,10,11,12,13,14,15);
  return __builtin_bit_cast(v16bf, c);
}
// A-matrix (16x32, M x K): lane = 16*half + m; elems 0..7 -> K=8*half+e, 8..15 -> K=16+8*half+e.
static __device__ __forceinline__ v16bf load_a(const bf16* row_k0, int hlf) {
  return frag2(row_k0 + 8 * hlf, row_k0 + 16 + 8 * hlf);
}
// B-matrix (32x16, K x N): lane = 16*half + n; elems 0..7 -> K=8*half+e, 8..15 -> K=16+8*half+e.
static __device__ __forceinline__ v16bf load_b(const bf16* col_k0, int hlf) {
  return frag2(col_k0 + 8 * hlf, col_k0 + 16 + 8 * hlf);
}
#define WMMA_BF16(a, b, c) \
  __builtin_amdgcn_wmma_f32_16x16x32_bf16(false, (a), false, (b), (short)0, (c), false, false)

// 32-bit LDS byte offset of a shared-memory pointer (generic -> addrspace(3)).
typedef __attribute__((address_space(3))) const void as3_void;
static __device__ __forceinline__ unsigned lds_off(const void* p) {
  return (unsigned)(size_t)(as3_void*)p;
}

// Tensor Data Mover: load a 64-row x 32-col bf16 tile (row stride = K elements)
// from global into LDS, padding each 64B row with 16B -> LDS rows of 40 elements.
// D# built per CDNA5 ISA 8.3/8.4; issued per-wave (EXEC ignored), TENSORcnt tracked.
static __device__ __forceinline__ void tdm_load_tile(const bf16* gsrc, unsigned lds_byte,
                                                     int K) {
  unsigned long long ga = (unsigned long long)(size_t)gsrc;
  u32x4 g0;
  g0[0] = 1u;                                       // count=1 (valid), user mode
  g0[1] = lds_byte;                                 // lds_addr
  g0[2] = (unsigned)(ga & 0xFFFFFFFFu);             // global_addr[31:0]
  g0[3] = (unsigned)((ga >> 32) & 0x01FFFFFFu)      // global_addr[56:32]
          | (2u << 30);                             // type = 2 ("image")
  i32x8 g1;
  g1[0] = (int)((1u << 16)                          // data_size = 2 bytes
                | (1u << 20)                        // pad_enable
                | (3u << 22)                        // pad_interval: 16 dwords
                | (3u << 25));                      // pad_amount:   4 dwords
  g1[1] = (int)(((unsigned)K & 0xFFFFu) << 16);     // tensor_dim0[15:0] = K
  g1[2] = (int)((4096u & 0xFFFFu) << 16);           // tensor_dim0 hi=0 | tensor_dim1[15:0]=4096
  g1[3] = (int)(32u << 16);                         // tensor_dim1 hi=0 | tile_dim0 = 32
  g1[4] = 64;                                       // tile_dim1 = 64 | tile_dim2 = 0
  g1[5] = K;                                        // tensor_dim0_stride[31:0] = K
  g1[6] = 0;                                        // stride hi | tensor_dim1_stride lo
  g1[7] = 0;
  i32x4 gz4 = {0, 0, 0, 0};                         // groups 2/3 unused (2-D tensor)
  i32x8 gz8 = {0, 0, 0, 0, 0, 0, 0, 0};             // trailing group (6-arg toolchain form)
  __builtin_amdgcn_tensor_load_to_lds(g0, g1, gz4, gz4, gz8, 0);
}

// ---------------------------------------------------------------- utilities
__global__ void f32_to_bf16(const float* __restrict__ in, bf16* __restrict__ out, int n) {
  int i = blockIdx.x * 256 + threadIdx.x;
  if (i < n) out[i] = (bf16)in[i];
}

__global__ __launch_bounds__(256) void layernorm_bf16(
    const float* __restrict__ x, const float* __restrict__ g, const float* __restrict__ b,
    bf16* __restrict__ out) {
  int row = blockIdx.x;
  const float* xr = x + (size_t)row * 1024;
  __shared__ float red[256];
  float s = 0.f;
  for (int i = threadIdx.x; i < 1024; i += 256) s += xr[i];
  red[threadIdx.x] = s; __syncthreads();
  for (int st = 128; st > 0; st >>= 1) {
    if (threadIdx.x < st) red[threadIdx.x] += red[threadIdx.x + st];
    __syncthreads();
  }
  float mean = red[0] * (1.f / 1024.f);
  __syncthreads();
  float v = 0.f;
  for (int i = threadIdx.x; i < 1024; i += 256) { float d = xr[i] - mean; v += d * d; }
  red[threadIdx.x] = v; __syncthreads();
  for (int st = 128; st > 0; st >>= 1) {
    if (threadIdx.x < st) red[threadIdx.x] += red[threadIdx.x + st];
    __syncthreads();
  }
  float rstd = rsqrtf(red[0] * (1.f / 1024.f) + 1e-5f);
  bf16* orow = out + (size_t)row * 1024;
  for (int i = threadIdx.x; i < 1024; i += 256)
    orow[i] = (bf16)((xr[i] - mean) * rstd * g[i] + b[i]);
}

// ------------------------------------------------------------- WMMA GEMM
// y[m,n] = sum_k A[m,k] * W[n,k] + bias[n]; block tile 64(M) x 128(N), 8 waves,
// wave w owns 16 N-columns; K stepped by 32; A staged via TDM (tensor_load_to_lds).
// MODE: 1 = bf16 out (scaled), 2 = bf16 transposed per-batch (vT[b][n][i]),
//       3 = f32 out + residual, 4 = relu -> bf16 out.
template <int MODE>
__global__ __launch_bounds__(256) void gemm_wmma(
    const bf16* __restrict__ A, const bf16* __restrict__ W,
    const float* __restrict__ bias, float scale,
    const float* __restrict__ resid,
    float* __restrict__ outF, bf16* __restrict__ outB,
    int Nout, int K) {
  __shared__ __align__(128) bf16 Asub[64][40];  // 64x32 tile, TDM pads rows 64B -> 80B
  const int wave = threadIdx.x >> 5;
  const int lane = threadIdx.x & 31;
  const int hlf  = lane >> 4;
  const int lm   = lane & 15;
  const int m0 = blockIdx.x * 64;
  const int n0 = blockIdx.y * 128 + wave * 16;
  const unsigned lds_base = lds_off(&Asub[0][0]);
  v8f acc[4] = {};
  const bf16* wcol = W + (size_t)(n0 + lm) * K;
  for (int k0 = 0; k0 < K; k0 += 32) {
    __syncthreads();                       // all waves done reading previous tile
    if (threadIdx.x < 32) {                // wave 0 drives the DMA (per-wave op)
      tdm_load_tile(A + (size_t)m0 * K + k0, lds_base, K);
      __builtin_amdgcn_s_wait_tensorcnt(0);
    }
    __syncthreads();                       // tile visible to all waves
    v16bf bfr = load_b(wcol + k0, hlf);
    if (k0 + 32 < K) __builtin_prefetch(wcol + k0 + 32, 0, 0);
#pragma unroll
    for (int t = 0; t < 4; t++) {
      v16bf afr = load_a(&Asub[t * 16 + lm][0], hlf);
      acc[t] = WMMA_BF16(afr, bfr, acc[t]);
    }
  }
  float bv = bias[n0 + lm];
#pragma unroll
  for (int t = 0; t < 4; t++) {
#pragma unroll
    for (int r = 0; r < 8; r++) {
      int gm = m0 + t * 16 + r + 8 * hlf;  // C layout: lane = 16*half + n, VGPR r -> M = r + 8*half
      int gn = n0 + lm;
      float v = acc[t][r] + bv;
      if (MODE == 1) {
        outB[(size_t)gm * Nout + gn] = (bf16)(v * scale);
      } else if (MODE == 2) {
        int bb = gm >> 10, ib = gm & 1023;
        outB[((size_t)bb * 512 + gn) * 1024 + ib] = (bf16)v;  // vT[b][n][i]
      } else if (MODE == 3) {
        outF[(size_t)gm * Nout + gn] = v + resid[(size_t)gm * Nout + gn];
      } else {
        outB[(size_t)gm * Nout + gn] = (bf16)fmaxf(v, 0.f);
      }
    }
  }
}

// ------------------------------------------------ fused compositional attention
// One block = one (b, h, 16-query-row tile). 8 waves.
// Pass 1: row-max of scores (WMMA). Pass 2: exp -> P (bf16 in LDS) + row sums.
// Pass 3: wave w computes PV for rule r=w (64 cols), folds 1/rowsum, does the
// rule-selection dot with Ws[32+d], softmax over R, weighted sum via LDS atomics.
// (qv branch dropped: constant across R => cancels in the softmax over R.)
__global__ __launch_bounds__(256) void attn_fused(
    const bf16* __restrict__ q, const bf16* __restrict__ k, const bf16* __restrict__ vT,
    const float* __restrict__ Ws, bf16* __restrict__ attn_out) {
  const int it = blockIdx.x & 63;
  const int bh = blockIdx.x >> 6;
  const int b  = bh >> 4;
  const int h  = bh & 15;
  const int i0 = b * 1024 + it * 16;
  const int wave = threadIdx.x >> 5;
  const int lane = threadIdx.x & 31;
  const int hlf  = lane >> 4;
  const int lm   = lane & 15;

  __shared__ __align__(128) bf16 P[16][1032];  // exp(scores), bf16, padded rows
  __shared__ float redmax[8][16];
  __shared__ float redsum[8][16];
  __shared__ float rowmax[16];
  __shared__ float inv_sum[16];
  __shared__ float compv[8][16];
  __shared__ float wgt[8][16];
  __shared__ float oacc[16][64];

  for (int i = threadIdx.x; i < 16 * 64; i += 256) ((float*)oacc)[i] = 0.f;

  const bf16* qrow = q + (size_t)(i0 + lm) * 1024 + h * 64;  // q pre-scaled by 1/8
  v16bf qa0 = load_a(qrow, hlf);
  v16bf qa1 = load_a(qrow + 32, hlf);

  // ---- pass 1: row max (uniform trip count -> no EXEC-mask loop)
  float rmx[8];
#pragma unroll
  for (int r = 0; r < 8; r++) rmx[r] = -3.0e38f;
#pragma unroll 1
  for (int jj = 0; jj < 8; ++jj) {
    int jt = jj * 8 + wave;
    const bf16* krow = k + (size_t)(b * 1024 + jt * 16 + lm) * 1024 + h * 64;
    v16bf kb0 = load_b(krow, hlf);
    v16bf kb1 = load_b(krow + 32, hlf);
    v8f c = {};
    c = WMMA_BF16(qa0, kb0, c);
    c = WMMA_BF16(qa1, kb1, c);
#pragma unroll
    for (int r = 0; r < 8; r++) rmx[r] = fmaxf(rmx[r], c[r]);
  }
#pragma unroll
  for (int off = 1; off < 16; off <<= 1)
#pragma unroll
    for (int r = 0; r < 8; r++) rmx[r] = fmaxf(rmx[r], __shfl_xor(rmx[r], off, 16));
  if (lm == 0)
    for (int r = 0; r < 8; r++) redmax[wave][r + 8 * hlf] = rmx[r];
  __syncthreads();
  if (threadIdx.x < 16) {
    float mv = -3.0e38f;
    for (int w = 0; w < 8; w++) mv = fmaxf(mv, redmax[w][threadIdx.x]);
    rowmax[threadIdx.x] = mv;
  }
  __syncthreads();

  // ---- pass 2: exp + store P + row sums
  float rmax[8];
#pragma unroll
  for (int r = 0; r < 8; r++) rmax[r] = rowmax[r + 8 * hlf];
  float rsum[8] = {0.f, 0.f, 0.f, 0.f, 0.f, 0.f, 0.f, 0.f};
#pragma unroll 1
  for (int jj = 0; jj < 8; ++jj) {
    int jt = jj * 8 + wave;
    int j0 = jt * 16;
    const bf16* krow = k + (size_t)(b * 1024 + j0 + lm) * 1024 + h * 64;
    v16bf kb0 = load_b(krow, hlf);
    v16bf kb1 = load_b(krow + 32, hlf);
    v8f c = {};
    c = WMMA_BF16(qa0, kb0, c);
    c = WMMA_BF16(qa1, kb1, c);
#pragma unroll
    for (int r = 0; r < 8; r++) {
      float p = __expf(c[r] - rmax[r]);
      rsum[r] += p;
      P[r + 8 * hlf][j0 + lm] = (bf16)p;
    }
  }
#pragma unroll
  for (int off = 1; off < 16; off <<= 1)
#pragma unroll
    for (int r = 0; r < 8; r++) rsum[r] += __shfl_xor(rsum[r], off, 16);
  if (lm == 0)
    for (int r = 0; r < 8; r++) redsum[wave][r + 8 * hlf] = rsum[r];
  __syncthreads();
  if (threadIdx.x < 16) {
    float sv = 0.f;
    for (int w = 0; w < 8; w++) sv += redsum[w][threadIdx.x];
    inv_sum[threadIdx.x] = 1.f / sv;
  }
  __syncthreads();

  // ---- pass 3: PV for this wave's rule slice (cols wave*64 .. +63)
  v8f pv[4] = {};
  const int n0v = wave * 64;
  for (int j0 = 0; j0 < 1024; j0 += 32) {
    v16bf pa = load_a(&P[lm][j0], hlf);
#pragma unroll
    for (int t = 0; t < 4; t++) {
      const bf16* vrow = vT + (size_t)(b * 512 + n0v + t * 16 + lm) * 1024 + j0;
      v16bf vb = load_b(vrow, hlf);
      pv[t] = WMMA_BF16(pa, vb, pv[t]);
    }
  }
  float isv[8];
#pragma unroll
  for (int r = 0; r < 8; r++) isv[r] = inv_sum[r + 8 * hlf];
#pragma unroll
  for (int t = 0; t < 4; t++)
#pragma unroll
    for (int r = 0; r < 8; r++) pv[t][r] *= isv[r];

  // rule-selection score: sum_d Ws[32+d] * out[m, r=wave, d]
  float wsv[4];
#pragma unroll
  for (int t = 0; t < 4; t++) wsv[t] = Ws[32 + t * 16 + lm];
  float part[8];
#pragma unroll
  for (int r = 0; r < 8; r++) {
    float s = 0.f;
#pragma unroll
    for (int t = 0; t < 4; t++) s += wsv[t] * pv[t][r];
    part[r] = s;
  }
#pragma unroll
  for (int off = 1; off < 16; off <<= 1)
#pragma unroll
    for (int r = 0; r < 8; r++) part[r] += __shfl_xor(part[r], off, 16);
  if (lm == 0)
    for (int r = 0; r < 8; r++) compv[wave][r + 8 * hlf] = part[r];
  __syncthreads();

  if (threadIdx.x < 16) {
    int m = threadIdx.x;
    float mx = -3.0e38f, cv[8];
    for (int r = 0; r < 8; r++) { cv[r] = compv[r][m]; mx = fmaxf(mx, cv[r]); }
    float s = 0.f;
    for (int r = 0; r < 8; r++) { cv[r] = __expf(cv[r] - mx); s += cv[r]; }
    float inv = 1.f / s;
    for (int r = 0; r < 8; r++) wgt[r][m] = cv[r] * inv;
  }
  __syncthreads();

  float wg[8];
#pragma unroll
  for (int r = 0; r < 8; r++) wg[r] = wgt[wave][r + 8 * hlf];
#pragma unroll
  for (int t = 0; t < 4; t++)
#pragma unroll
    for (int r = 0; r < 8; r++)
      atomicAdd(&oacc[r + 8 * hlf][t * 16 + lm], wg[r] * pv[t][r]);
  __syncthreads();

  for (int i = threadIdx.x; i < 16 * 64; i += 256) {
    int m = i >> 6, d = i & 63;
    attn_out[(size_t)(i0 + m) * 1024 + h * 64 + d] = (bf16)oacc[m][d];
  }
}

// ---------------------------------------------------------------- driver
extern "C" void kernel_launch(void* const* d_in, const int* in_sizes, int n_in,
                              void* d_out, int out_size, void* d_ws, size_t ws_size,
                              hipStream_t stream) {
  (void)in_sizes; (void)n_in; (void)out_size; (void)ws_size;
  const float* x   = (const float*)d_in[0];
  const float* Wq  = (const float*)d_in[1];
  const float* bq  = (const float*)d_in[2];
  const float* Wk  = (const float*)d_in[3];
  const float* bk  = (const float*)d_in[4];
  const float* Wv  = (const float*)d_in[5];
  const float* bv  = (const float*)d_in[6];
  // d_in[7] Wqv, d_in[8] bqv, d_in[10] bs: provably dead (constant across R
  // before the softmax over R => shift-invariant, contributes nothing).
  const float* Ws  = (const float*)d_in[9];
  const float* Wf  = (const float*)d_in[11];
  const float* bfb = (const float*)d_in[12];
  const float* Wr1 = (const float*)d_in[13];
  const float* br1 = (const float*)d_in[14];
  const float* Wr2 = (const float*)d_in[15];
  const float* br2 = (const float*)d_in[16];
  const float* g1  = (const float*)d_in[17];
  const float* b1  = (const float*)d_in[18];
  const float* g2  = (const float*)d_in[19];
  const float* b2  = (const float*)d_in[20];

  char* ws = (char*)d_ws;
  const size_t MB = 1ull << 20;
  bf16* Wq_b   = (bf16*)(ws + 0 * MB);    // 2 MB
  bf16* Wk_b   = (bf16*)(ws + 2 * MB);    // 2 MB
  bf16* Wv_b   = (bf16*)(ws + 4 * MB);    // 1 MB
  bf16* Wf_b   = (bf16*)(ws + 5 * MB);    // 2 MB
  bf16* Wr1_b  = (bf16*)(ws + 7 * MB);    // 4 MB
  bf16* Wr2_b  = (bf16*)(ws + 11 * MB);   // 4 MB
  bf16* xn_b   = (bf16*)(ws + 15 * MB);   // 8 MB   LN1(x) in bf16
  bf16* q_b    = (bf16*)(ws + 23 * MB);   // 8 MB   (scaled by 1/8)
  bf16* k_b    = (bf16*)(ws + 31 * MB);   // 8 MB
  bf16* vT_b   = (bf16*)(ws + 39 * MB);   // 4 MB   vT[b][512][1024]
  bf16* attn_b = (bf16*)(ws + 43 * MB);   // 8 MB
  float* res2  = (float*)(ws + 51 * MB);  // 16 MB
  bf16* on_b   = (bf16*)(ws + 23 * MB);   // alias q_b   (dead after attention)
  bf16* h_b    = (bf16*)(ws + 31 * MB);   // alias k/vT/attn (dead after Wf GEMM), 16 MB

  auto cvt = [&](const float* src, bf16* dst, int n) {
    f32_to_bf16<<<dim3((n + 255) / 256), dim3(256), 0, stream>>>(src, dst, n);
  };
  cvt(Wq, Wq_b, 1024 * 1024);
  cvt(Wk, Wk_b, 1024 * 1024);
  cvt(Wv, Wv_b, 512 * 1024);
  cvt(Wf, Wf_b, 1024 * 1024);
  cvt(Wr1, Wr1_b, 2048 * 1024);
  cvt(Wr2, Wr2_b, 1024 * 2048);

  layernorm_bf16<<<4096, 256, 0, stream>>>(x, g1, b1, xn_b);

  gemm_wmma<1><<<dim3(64, 8),  256, 0, stream>>>(xn_b, Wq_b, bq, 0.125f, nullptr, nullptr, q_b, 1024, 1024);
  gemm_wmma<1><<<dim3(64, 8),  256, 0, stream>>>(xn_b, Wk_b, bk, 1.0f,   nullptr, nullptr, k_b, 1024, 1024);
  gemm_wmma<2><<<dim3(64, 4),  256, 0, stream>>>(xn_b, Wv_b, bv, 1.0f,   nullptr, nullptr, vT_b, 512, 1024);

  attn_fused<<<4096, 256, 0, stream>>>(q_b, k_b, vT_b, Ws, attn_b);

  gemm_wmma<3><<<dim3(64, 8),  256, 0, stream>>>(attn_b, Wf_b, bfb, 1.0f, x, res2, nullptr, 1024, 1024);

  layernorm_bf16<<<4096, 256, 0, stream>>>(res2, g2, b2, on_b);

  gemm_wmma<4><<<dim3(64, 16), 256, 0, stream>>>(on_b, Wr1_b, br1, 1.0f, nullptr, nullptr, h_b, 2048, 1024);
  gemm_wmma<3><<<dim3(64, 8),  256, 0, stream>>>(h_b, Wr2_b, br2, 1.0f, res2, (float*)d_out, nullptr, 1024, 2048);
}
